// MaskedSelfAttention_76562087018941
// MI455X (gfx1250) — matrix operand order repaired and checked by reading, not compiled, not run on hardware
//
#include <hip/hip_runtime.h>

// ---------------------------------------------------------------------------
// MI455X (gfx1250) masked self-attention, bf16 WMMA + TDM-staged LDS.
// Pipeline:
//   xbf = bf16(x); wT* = bf16(W^T)  (one-shot conversion kernels)
//   q = (xbf@wq + bq)/8  -> bf16 [M,D]        (WMMA GEMM, TDM-staged LDS)
//   k =  xbf@wk + bk     -> bf16 [M,D]
//   v =  xbf@wv + bv     -> bf16 vt[b,h,d,s]  (head-transposed epilogue)
//   flash attention (additive -1e9 on k<=q, per reference tril) -> bf16 [M,D]
//   out = ao@wo + bo     -> f32 d_out
// ---------------------------------------------------------------------------

typedef __bf16 v16bf __attribute__((ext_vector_type(16)));
typedef __bf16 v8bf  __attribute__((ext_vector_type(8)));
typedef float  v8f   __attribute__((ext_vector_type(8)));
typedef float  v4f   __attribute__((ext_vector_type(4)));
typedef unsigned int u32x4 __attribute__((ext_vector_type(4)));
typedef int          i32x4 __attribute__((ext_vector_type(4)));
typedef int          i32x8 __attribute__((ext_vector_type(8)));

#define AS3 __attribute__((address_space(3)))

#if __has_builtin(__builtin_amdgcn_tensor_load_to_lds) && \
    __has_builtin(__builtin_amdgcn_s_wait_tensorcnt)
#define HAVE_TDM 1
#else
#define HAVE_TDM 0
#endif

__device__ __forceinline__ v16bf mk16(v8bf lo, v8bf hi) {
  return __builtin_shufflevector(lo, hi, 0,1,2,3,4,5,6,7,8,9,10,11,12,13,14,15);
}

__device__ __forceinline__ v8f wmma_bf16(v16bf a, v16bf b, v8f c) {
  return __builtin_amdgcn_wmma_f32_16x16x32_bf16(false, a, false, b,
                                                 (short)0, c, false, false);
}

#if HAVE_TDM
__device__ __forceinline__ unsigned ldsOffset(void* p) {
  return (unsigned)(unsigned long long)(AS3 char*)(char*)p;
}
// 2D tile load global->LDS via Tensor Data Mover. D# packing per CDNA5 ISA
// ch.8: group0 = {count/lds_addr/global_addr/type}, group1 = {flags, dims}.
// This toolchain's builtin is the 6-arg form: (g0, g1, g2, g3, g4, cpol);
// groups 2..4 are unused for a 2D tile (tile_dim2 = 0) -> zeros.
__device__ __forceinline__ void tdm_load_2d(unsigned ldsAddr, const void* gptr,
                                            unsigned tileRows, unsigned tileCols,
                                            unsigned strideElems,
                                            unsigned tensorRows, unsigned tensorCols)
{
  unsigned long long ga = (unsigned long long)gptr;
  u32x4 g0;
  g0[0] = 1u;                                  // count=1, user mode, no gather
  g0[1] = ldsAddr;                             // lds_addr [63:32]
  g0[2] = (unsigned)(ga & 0xFFFFFFFFu);        // global_addr [95:64]
  g0[3] = (unsigned)((ga >> 32) & 0x01FFFFFFu) // global_addr [120:96]
        | (2u << 30);                          // type=2 ("image") [127:126]
  i32x8 g1;
  g1[0] = (int)(1u << 16);                     // wg_mask=0, data_size=1 (2B)
  g1[1] = (int)((tensorCols & 0xFFFFu) << 16);                 // dim0 lo16
  g1[2] = (int)(((tensorCols >> 16) & 0xFFFFu) |
                ((tensorRows & 0xFFFFu) << 16));               // dim0 hi / dim1 lo
  g1[3] = (int)(((tensorRows >> 16) & 0xFFFFu) |
                ((tileCols & 0xFFFFu) << 16));                 // dim1 hi / tile0
  g1[4] = (int)(tileRows & 0xFFFFu);                           // tile1 (tile2=0)
  g1[5] = (int)strideElems;                                    // dim0 stride lo32
  g1[6] = 0;                                                   // stride hi / d1s
  g1[7] = 0;
  i32x4 z4 = {0, 0, 0, 0};
  i32x8 z8 = {0, 0, 0, 0, 0, 0, 0, 0};
  __builtin_amdgcn_tensor_load_to_lds(g0, g1, z4, z4, z8, 0);
}
#endif

// ---------------------------------------------------------------------------
// Conversion kernels (one-shot, bandwidth trivial vs 23.3 TB/s)
// ---------------------------------------------------------------------------
__global__ __launch_bounds__(256) void cvt_bf16(const float* __restrict__ in,
                                                __bf16* __restrict__ out) {
  const size_t i = ((size_t)blockIdx.x * 256 + threadIdx.x) * 4;
  v4f f = *(const v4f*)(in + i);
  out[i + 0] = (__bf16)f.x;
  out[i + 1] = (__bf16)f.y;
  out[i + 2] = (__bf16)f.z;
  out[i + 3] = (__bf16)f.w;
}

// W[K,N] f32 -> WT[N,K] bf16 (coalesced read; scattered bf16 write is fine:
// 4 MB per weight, noise at this bandwidth)
__global__ __launch_bounds__(256) void cvt_T_bf16(const float* __restrict__ W,
                                                  __bf16* __restrict__ WT,
                                                  int K, int N) {
  const size_t i4 = ((size_t)blockIdx.x * 256 + threadIdx.x) * 4;
  const int k = (int)(i4 / N);
  const int n = (int)(i4 % N);
  v4f f = *(const v4f*)(W + i4);
  WT[(size_t)(n + 0) * K + k] = (__bf16)f.x;
  WT[(size_t)(n + 1) * K + k] = (__bf16)f.y;
  WT[(size_t)(n + 2) * K + k] = (__bf16)f.z;
  WT[(size_t)(n + 3) * K + k] = (__bf16)f.w;
}

// ---------------------------------------------------------------------------
// GEMM: C[M,N] = (A[M,K]bf16 @ WT[N,K]bf16^T + bias) * scale
// Block 256 threads (8 waves), tile 128x128x32, double-buffered LDS with
// TDM staging (wave 0 issues two descriptors, TENSORcnt + barrier handoff).
// OUTMODE: 0 = bf16 [M,N] ; 1 = bf16 vt[b,h,d,s] ; 2 = f32 [M,N]
// ---------------------------------------------------------------------------
template <int OUTMODE>
__global__ __launch_bounds__(256) void gemm_wmma(
    const __bf16* __restrict__ A, const __bf16* __restrict__ WT,
    const float* __restrict__ bias, void* __restrict__ Cout,
    int M, int N, int K, float scale)
{
  __shared__ __bf16 sA[2][128 * 32];   // [row][k], 8 KB per buffer
  __shared__ __bf16 sW[2][128 * 32];   // [n][k]

  const int t    = threadIdx.x;
  const int lane = t & 31;
  const int wave = t >> 5;
  const int wm   = wave >> 1;
  const int wn   = wave & 1;
  const int l16  = lane & 15;
  const int half = lane >> 4;
  const int m0   = blockIdx.x * 128;
  const int n0   = blockIdx.y * 128;

  v8f acc[2][4] = {};

  auto stage = [&](int buf, int k0) {
#if HAVE_TDM
    if (wave == 0) {
      tdm_load_2d(ldsOffset(&sA[buf][0]), A  + (size_t)m0 * K + k0,
                  128, 32, (unsigned)K, (unsigned)M, (unsigned)K);
      tdm_load_2d(ldsOffset(&sW[buf][0]), WT + (size_t)n0 * K + k0,
                  128, 32, (unsigned)K, (unsigned)N, (unsigned)K);
    }
#else
    const int row = t >> 1;
    const int c0  = (t & 1) * 16;
    *(v8bf*)&sA[buf][row * 32 + c0]     = *(const v8bf*)(A  + (size_t)(m0 + row) * K + k0 + c0);
    *(v8bf*)&sA[buf][row * 32 + c0 + 8] = *(const v8bf*)(A  + (size_t)(m0 + row) * K + k0 + c0 + 8);
    *(v8bf*)&sW[buf][row * 32 + c0]     = *(const v8bf*)(WT + (size_t)(n0 + row) * K + k0 + c0);
    *(v8bf*)&sW[buf][row * 32 + c0 + 8] = *(const v8bf*)(WT + (size_t)(n0 + row) * K + k0 + c0 + 8);
#endif
  };
  auto stageWait = [&]() {
#if HAVE_TDM
    if (wave == 0) __builtin_amdgcn_s_wait_tensorcnt(0);
#endif
  };

  stage(0, 0);
  stageWait();
  __syncthreads();

  for (int k0 = 0; k0 < K; k0 += 32) {
    const int buf = (k0 >> 5) & 1;
    if (k0 + 32 < K) stage(buf ^ 1, k0 + 32);   // DMA next tile during compute

    // A fragment: lane half 0 -> K 0..15, half 1 -> K 16..31 (contiguous)
    v16bf af[2];
    {
      const int kbA = half * 16;
#pragma unroll
      for (int tm = 0; tm < 2; ++tm) {
        const int r = wm * 32 + tm * 16 + l16;
        v8bf lo = *(const v8bf*)(&sA[buf][r * 32 + kbA]);
        v8bf hi = *(const v8bf*)(&sA[buf][r * 32 + kbA + 8]);
        af[tm] = mk16(lo, hi);
      }
    }
    // B fragment: lane half 0 -> K 0..15, half 1 -> K 16..31 (contiguous)
    v16bf bfr[4];
    {
      const int kbB = half * 16;
#pragma unroll
      for (int tn = 0; tn < 4; ++tn) {
        const int n = wn * 64 + tn * 16 + l16;
        v8bf lo = *(const v8bf*)(&sW[buf][n * 32 + kbB]);
        v8bf hi = *(const v8bf*)(&sW[buf][n * 32 + kbB + 8]);
        bfr[tn] = mk16(lo, hi);
      }
    }
#pragma unroll
    for (int tm = 0; tm < 2; ++tm)
#pragma unroll
      for (int tn = 0; tn < 4; ++tn)
        acc[tm][tn] = wmma_bf16(af[tm], bfr[tn], acc[tm][tn]);

    stageWait();
    __syncthreads();
  }

  // epilogue: lane holds N=l16, rows j + 8*half
#pragma unroll
  for (int tm = 0; tm < 2; ++tm) {
#pragma unroll
    for (int tn = 0; tn < 4; ++tn) {
      const int gn = n0 + wn * 64 + tn * 16 + l16;
      const float bv = bias[gn];
      const int rbase = m0 + wm * 32 + tm * 16 + half * 8;
#pragma unroll
      for (int j = 0; j < 8; ++j) {
        const float val = (acc[tm][tn][j] + bv) * scale;
        const int m = rbase + j;
        if (OUTMODE == 0) {
          ((__bf16*)Cout)[(size_t)m * N + gn] = (__bf16)val;
        } else if (OUTMODE == 2) {
          ((float*)Cout)[(size_t)m * N + gn] = val;
        } else {
          // vt[b, h, d, s]: b=m/2048, s=m%2048, h=gn/64, d=gn%64
          const int b = m >> 11, s = m & 2047;
          const int h = gn >> 6, d = gn & 63;
          ((__bf16*)Cout)[((size_t)((b << 4) + h) * 64 + d) * 2048 + s] = (__bf16)val;
        }
      }
    }
  }
}

// ---------------------------------------------------------------------------
// Flash attention. Grid (B*H, S/64), block 128 (4 waves), wave = 16 q-rows.
// No inter-wave sync at all: K and V fragments come straight from global
// (V pre-transposed to vt[b,h,d,s]); only LDS use is the wave-private P
// shuffle (C-layout -> A-layout for the P@V WMMA).
// ---------------------------------------------------------------------------
__global__ __launch_bounds__(128) void flash_attn(
    const __bf16* __restrict__ Q, const __bf16* __restrict__ Kmat,
    const __bf16* __restrict__ VT, __bf16* __restrict__ AO)
{
  constexpr int S = 2048, D = 1024, HD = 64, H = 16;
  __shared__ __bf16 P[4][16 * 32];

  const int t    = threadIdx.x;
  const int lane = t & 31;
  const int wave = t >> 5;
  const int l16  = lane & 15;
  const int half = lane >> 4;
  const int b    = blockIdx.x / H;
  const int h    = blockIdx.x % H;
  const int q0   = blockIdx.y * 64 + wave * 16;

  v16bf qf[2];
  {
    const __bf16* base = Q + (size_t)(b * S + q0 + l16) * D + h * HD;
    const int kbA = half * 16;
#pragma unroll
    for (int ds = 0; ds < 2; ++ds) {
      v8bf lo = *(const v8bf*)(base + ds * 32 + kbA);
      v8bf hi = *(const v8bf*)(base + ds * 32 + kbA + 8);
      qf[ds] = mk16(lo, hi);
    }
  }

  float mrow[8], lrow[8];
  v8f acc[4] = {};
#pragma unroll
  for (int j = 0; j < 8; ++j) { mrow[j] = -3.0e38f; lrow[j] = 0.0f; }

  const __bf16* kBase = Kmat + (size_t)b * S * D + h * HD;
  const __bf16* vBase = VT + (size_t)((b << 4) + h) * HD * S;

  for (int kb = 0; kb < S; kb += 32) {
    if (kb + 32 < S) {  // prefetch next key block (global_prefetch_b8)
      __builtin_prefetch(kBase + (size_t)(kb + 32 + lane) * D, 0, 1);
      __builtin_prefetch(vBase + (size_t)(lane * 2) * S + kb + 32, 0, 1);
    }

    // ---- scores: 2 n-tiles x 2 depth steps; K frags straight from global ----
    v8f st[2] = {};
#pragma unroll
    for (int nt = 0; nt < 2; ++nt) {
#pragma unroll
      for (int ds = 0; ds < 2; ++ds) {
        const __bf16* kp = kBase + (size_t)(kb + nt * 16 + l16) * D
                           + ds * 32 + half * 16;
        v8bf lo = *(const v8bf*)(kp);
        v8bf hi = *(const v8bf*)(kp + 8);
        st[nt] = wmma_bf16(qf[ds], mk16(lo, hi), st[nt]);
      }
    }

    // ---- additive tril mask (k <= q gets -1e9), faithful to reference ----
#pragma unroll
    for (int nt = 0; nt < 2; ++nt) {
      const int col = kb + nt * 16 + l16;
#pragma unroll
      for (int j = 0; j < 8; ++j) {
        const int row = q0 + half * 8 + j;
        if (col <= row) st[nt][j] += -1.0e9f;
      }
    }

    // ---- online softmax (rows live in 16-lane groups) ----
    float alpha[8];
#pragma unroll
    for (int j = 0; j < 8; ++j) {
      float mx = fmaxf(st[0][j], st[1][j]);
      mx = fmaxf(mx, __shfl_xor(mx, 1, 32));
      mx = fmaxf(mx, __shfl_xor(mx, 2, 32));
      mx = fmaxf(mx, __shfl_xor(mx, 4, 32));
      mx = fmaxf(mx, __shfl_xor(mx, 8, 32));
      const float mnew = fmaxf(mrow[j], mx);
      const float p0 = __expf(st[0][j] - mnew);
      const float p1 = __expf(st[1][j] - mnew);
      float rs = p0 + p1;
      rs += __shfl_xor(rs, 1, 32);
      rs += __shfl_xor(rs, 2, 32);
      rs += __shfl_xor(rs, 4, 32);
      rs += __shfl_xor(rs, 8, 32);
      alpha[j] = __expf(mrow[j] - mnew);
      lrow[j] = lrow[j] * alpha[j] + rs;
      mrow[j] = mnew;
      st[0][j] = p0;
      st[1][j] = p1;
    }

    // ---- P (C layout) -> wave-private LDS -> A layout fragment ----
    __bf16* pw = P[wave];
#pragma unroll
    for (int j = 0; j < 8; ++j) {
      const int row = half * 8 + j;
      pw[row * 32 + l16]      = (__bf16)st[0][j];
      pw[row * 32 + 16 + l16] = (__bf16)st[1][j];
    }

#pragma unroll
    for (int tn = 0; tn < 4; ++tn)
#pragma unroll
      for (int j = 0; j < 8; ++j) acc[tn][j] *= alpha[j];

    v16bf pf;
    {
      const int kbA = half * 16;
      v8bf lo = *(const v8bf*)(pw + l16 * 32 + kbA);
      v8bf hi = *(const v8bf*)(pw + l16 * 32 + kbA + 8);
      pf = mk16(lo, hi);
    }
    // ---- P @ V: V frags contiguous from vt[b,h,d,s] ----
#pragma unroll
    for (int tn = 0; tn < 4; ++tn) {
      const __bf16* vp = vBase + (size_t)(tn * 16 + l16) * S + kb + half * 16;
      v8bf lo = *(const v8bf*)(vp);
      v8bf hi = *(const v8bf*)(vp + 8);
      acc[tn] = wmma_bf16(pf, mk16(lo, hi), acc[tn]);
    }
  }

  // ---- bf16 merged-head output [B,S,D] ----
#pragma unroll
  for (int tn = 0; tn < 4; ++tn) {
    const int col = h * HD + tn * 16 + l16;
#pragma unroll
    for (int j = 0; j < 8; ++j) {
      const int s = q0 + half * 8 + j;
      AO[(size_t)(b * S + s) * D + col] = (__bf16)(acc[tn][j] / lrow[j]);
    }
  }
}

// ---------------------------------------------------------------------------
extern "C" void kernel_launch(void* const* d_in, const int* in_sizes, int n_in,
                              void* d_out, int out_size, void* d_ws, size_t ws_size,
                              hipStream_t stream) {
  (void)in_sizes; (void)n_in; (void)out_size; (void)ws_size;
  const float* x  = (const float*)d_in[0];
  // d_in[1] = mask: reference ignores it (tril applied unconditionally)
  const float* wq = (const float*)d_in[2];
  const float* bq = (const float*)d_in[3];
  const float* wk = (const float*)d_in[4];
  const float* bk = (const float*)d_in[5];
  const float* wv = (const float*)d_in[6];
  const float* bv = (const float*)d_in[7];
  const float* wo = (const float*)d_in[8];
  const float* bo = (const float*)d_in[9];
  float* out = (float*)d_out;

  constexpr int B = 2, S = 2048, D = 1024, H = 16;
  const int M = B * S;                        // 4096
  const size_t mdE = (size_t)M * D;           // 4M elems
  const size_t ddE = (size_t)D * D;           // 1M elems

  char* ws = (char*)d_ws;
  __bf16* xbf  = (__bf16*)(ws);                         //  8 MB
  __bf16* wqT  = (__bf16*)(ws + mdE * 2);               //  2 MB each
  __bf16* wkT  = wqT + ddE;
  __bf16* wvT  = wkT + ddE;
  __bf16* woT  = wvT + ddE;
  __bf16* q    = woT + ddE;                             //  8 MB
  __bf16* k    = q + mdE;                               //  8 MB
  __bf16* vt   = k + mdE;                               //  8 MB
  __bf16* ao   = vt + mdE;                              //  8 MB

  cvt_bf16<<<dim3((unsigned)(mdE / 1024)), dim3(256), 0, stream>>>(x, xbf);
  cvt_T_bf16<<<dim3((unsigned)(ddE / 1024)), dim3(256), 0, stream>>>(wq, wqT, D, D);
  cvt_T_bf16<<<dim3((unsigned)(ddE / 1024)), dim3(256), 0, stream>>>(wk, wkT, D, D);
  cvt_T_bf16<<<dim3((unsigned)(ddE / 1024)), dim3(256), 0, stream>>>(wv, wvT, D, D);
  cvt_T_bf16<<<dim3((unsigned)(ddE / 1024)), dim3(256), 0, stream>>>(wo, woT, D, D);

  const dim3 gGrid(M / 128, D / 128);
  const dim3 gBlk(256);
  gemm_wmma<0><<<gGrid, gBlk, 0, stream>>>(xbf, wqT, bq, q,  M, D, D, 0.125f);
  gemm_wmma<0><<<gGrid, gBlk, 0, stream>>>(xbf, wkT, bk, k,  M, D, D, 1.0f);
  gemm_wmma<1><<<gGrid, gBlk, 0, stream>>>(xbf, wvT, bv, vt, M, D, D, 1.0f);

  flash_attn<<<dim3(B * H, S / 64), dim3(128), 0, stream>>>(q, k, vt, ao);

  gemm_wmma<2><<<gGrid, gBlk, 0, stream>>>(ao, woT, bo, out, M, D, D, 1.0f);
}
